// BidirectionalAttention_24489903521818
// MI455X (gfx1250) — matrix operand order repaired and checked by reading, not compiled, not run on hardware
//
#include <hip/hip_runtime.h>
#include <hip/hip_bf16.h>

// ---------------------------------------------------------------------------
// Types for CDNA5 WMMA (wave32): V_WMMA_F32_16X16X32_BF16
// ---------------------------------------------------------------------------
typedef __bf16 bf16;
typedef __attribute__((ext_vector_type(16))) __bf16 v16bf;
typedef __attribute__((ext_vector_type(8)))  __bf16 v8bf;
typedef __attribute__((ext_vector_type(8)))  float  v8f;

union BV16 { v16bf v; v8bf h[2]; };

#define D_MODEL 2048
#define N_HEADS 16
#define HEAD_DIM 128
#define SEQ_T 2048
#define BATCH 2
#define NROWS (BATCH * SEQ_T)          // 4096 token rows
#define EPS 1e-6f
#define QK_SCALE 0.08838834764831845f  // 1/sqrt(128)

#define AS3 __attribute__((address_space(3)))

static __device__ __forceinline__ v8f wmma_bf16(const v16bf& a, const v16bf& b, v8f c) {
  return __builtin_amdgcn_wmma_f32_16x16x32_bf16(false, a, false, b, (short)0, c, false, false);
}

// LDS byte address (32-bit, addrspace(3)) for async-to-LDS instructions
static __device__ __forceinline__ unsigned lds_addr(const void* p) {
  return (unsigned)(uintptr_t)(AS3 const void*)p;
}

// CDNA5 async global->LDS copy, 16 bytes per lane (tracked by ASYNCcnt)
static __device__ __forceinline__ void async_load_b128(unsigned lds, const void* gaddr) {
  asm volatile("global_load_async_to_lds_b128 %0, %1, off"
               :: "v"(lds), "v"(gaddr) : "memory");
}
static __device__ __forceinline__ void wait_asynccnt0() {
  asm volatile("s_wait_asynccnt 0" ::: "memory");
}

// ---------------------------------------------------------------------------
// 1) f32 -> bf16 cast (weights)
// ---------------------------------------------------------------------------
__global__ void __launch_bounds__(256) cast_f32_bf16(const float* __restrict__ src,
                                                     bf16* __restrict__ dst, int n) {
  int i = blockIdx.x * 256 + threadIdx.x;
  int stride = gridDim.x * 256;
  for (; i < n; i += stride) dst[i] = (bf16)src[i];
}

// ---------------------------------------------------------------------------
// 2) RMSNorm: one block per token row, writes bf16 h
// ---------------------------------------------------------------------------
__global__ void __launch_bounds__(256) rmsnorm_kernel(const float* __restrict__ x,
                                                      const float* __restrict__ w,
                                                      bf16* __restrict__ h) {
  const int row = blockIdx.x;
  const int tid = threadIdx.x;
  const float* xr = x + (size_t)row * D_MODEL;
  float xv[8];
  float ss = 0.f;
#pragma unroll
  for (int it = 0; it < 8; ++it) {
    float v = xr[tid + 256 * it];
    xv[it] = v;
    ss += v * v;
  }
#pragma unroll
  for (int d = 16; d >= 1; d >>= 1) ss += __shfl_xor(ss, d, 32);
  __shared__ float red[8];
  if ((tid & 31) == 0) red[tid >> 5] = ss;
  __syncthreads();
  float tot = 0.f;
#pragma unroll
  for (int i = 0; i < 8; ++i) tot += red[i];
  float rstd = rsqrtf(tot / (float)D_MODEL + EPS);
  bf16* hr = h + (size_t)row * D_MODEL;
#pragma unroll
  for (int it = 0; it < 8; ++it) {
    int c = tid + 256 * it;
    hr[c] = (bf16)(xv[it] * rstd * w[c]);
  }
}

// ---------------------------------------------------------------------------
// 3) GEMM: C[M,N] = A[M,K] * W[N,K]^T  (bf16 in, f32 accum)
//    block tile 128x128, 8 waves of 64x32, K-step 32.
//    Double-buffered LDS staged via CDNA5 async global->LDS engine.
//    WRITE_BF16: bf16 C out ; else f32 (residual + C) out. Compile-time ->
//    straight-line epilogue (no per-element branches).
// ---------------------------------------------------------------------------
template <bool WRITE_BF16>
__global__ void __launch_bounds__(256) gemm_bt_kernel(const bf16* __restrict__ A,
                                                      const bf16* __restrict__ W,
                                                      int M, int N, int K,
                                                      bf16* __restrict__ outBf,
                                                      float* __restrict__ outF,
                                                      const float* __restrict__ residual) {
  __shared__ __align__(32) bf16 As[2][128 * 32];
  __shared__ __align__(32) bf16 Bs[2][128 * 32];

  const int tid  = threadIdx.x;
  const int wave = tid >> 5;
  const int lane = tid & 31;
  const int hi   = lane >> 4;        // half-wave select
  const int lr   = lane & 15;
  const int wm   = wave >> 2;        // 0..1 (M)
  const int wn   = wave & 3;         // 0..3 (N)
  const int m0   = blockIdx.y * 128;
  const int n0   = blockIdx.x * 128;

  // cooperative-load coordinates: each thread moves four 16B chunks per stage
  const int ldr = tid >> 2;          // 0..63
  const int ldc = (tid & 3) * 8;     // 0,8,16,24

  const bf16* gA0 = A + (size_t)(m0 + ldr) * K + ldc;
  const bf16* gA1 = A + (size_t)(m0 + ldr + 64) * K + ldc;
  const bf16* gB0 = W + (size_t)(n0 + ldr) * K + ldc;
  const bf16* gB1 = W + (size_t)(n0 + ldr + 64) * K + ldc;

  v8f acc[4][2];
#pragma unroll
  for (int i = 0; i < 4; ++i)
#pragma unroll
    for (int j = 0; j < 2; ++j) acc[i][j] = {};

  // async-stage one 128x32 A tile + 128x32 B tile into buffer `buf`
  auto stage = [&](int buf, int k0) {
    async_load_b128(lds_addr(&As[buf][ldr * 32 + ldc]),        gA0 + k0);
    async_load_b128(lds_addr(&As[buf][(ldr + 64) * 32 + ldc]), gA1 + k0);
    async_load_b128(lds_addr(&Bs[buf][ldr * 32 + ldc]),        gB0 + k0);
    async_load_b128(lds_addr(&Bs[buf][(ldr + 64) * 32 + ldc]), gB1 + k0);
  };

  // consume buffer `buf`: 8 WMMAs on the 64x32 wave tile
  auto compute = [&](int buf) {
    // A tiles: 16x32 bf16 (lane<16: K 0..15 ; lane>=16: K 16..31)
    BV16 a[4];
#pragma unroll
    for (int i = 0; i < 4; ++i) {
      const bf16* ap = &As[buf][(wm * 64 + i * 16 + lr) * 32];
      a[i].h[0] = *(const v8bf*)(ap + 16 * hi);
      a[i].h[1] = *(const v8bf*)(ap + 16 * hi + 8);
    }
#pragma unroll
    for (int jn = 0; jn < 2; ++jn) {
      // B tile 32x16: lane<16 -> col=lane,K0..15 ; lane>=16 -> col=lane-16,K16..31
      BV16 b;
      const bf16* bp = &Bs[buf][(wn * 32 + jn * 16 + lr) * 32 + 16 * hi];
      b.h[0] = *(const v8bf*)(bp);
      b.h[1] = *(const v8bf*)(bp + 8);
#pragma unroll
      for (int i = 0; i < 4; ++i) acc[i][jn] = wmma_bf16(a[i].v, b.v, acc[i][jn]);
    }
  };

  stage(0, 0);
  wait_asynccnt0();
  __syncthreads();

  // steady state: prefetch next tile async while computing current
  int k0 = 0;
  for (; k0 < K - 32; k0 += 32) {
    const int buf = (k0 >> 5) & 1;
    stage(buf ^ 1, k0 + 32);
    compute(buf);
    wait_asynccnt0();                // next tile resident before anyone flips buffers
    __syncthreads();
  }
  compute((k0 >> 5) & 1);            // last tile, no staging / no barrier

  // epilogue: C layout -> element (row = base + e + 8*hi, col = base + lr)
#pragma unroll
  for (int i = 0; i < 4; ++i)
#pragma unroll
    for (int jn = 0; jn < 2; ++jn)
#pragma unroll
      for (int e = 0; e < 8; ++e) {
        int row = m0 + wm * 64 + i * 16 + e + 8 * hi;
        int col = n0 + wn * 32 + jn * 16 + lr;
        size_t idx = (size_t)row * N + col;
        if constexpr (WRITE_BF16) outBf[idx] = (bf16)acc[i][jn][e];
        else                      outF[idx]  = residual[idx] + acc[i][jn][e];
      }
}

// ---------------------------------------------------------------------------
// 4) RoPE + split + transpose-V + fold 1/sqrt(hd) into Q
//    qkv:[B,T,3,H,128] bf16 -> Q,K:[B,H,T,128] ; Vt:[B,H,128,T]
// ---------------------------------------------------------------------------
__global__ void __launch_bounds__(256) rope_split_kernel(const bf16* __restrict__ qkv,
                                                         bf16* __restrict__ Q,
                                                         bf16* __restrict__ Kd,
                                                         bf16* __restrict__ Vt) {
  int idx = blockIdx.x * 256 + threadIdx.x;   // B*T*H*64 threads
  int d2 = idx & 63; int rest = idx >> 6;
  int h = rest & (N_HEADS - 1); rest >>= 4;
  int t = rest & (SEQ_T - 1);
  int b = rest >> 11;

  size_t qb = ((size_t)(b * SEQ_T + t)) * (3 * D_MODEL) + (size_t)h * HEAD_DIM;
  // angle: t * 10000^(-2*d2/128)  ->  exp(-d2 * ln(10000)/64)
  float invf = __expf(-0.14391156842389587f * (float)d2);
  float ang = (float)t * invf;
  float c = __cosf(ang), s = __sinf(ang);

  float q0 = (float)qkv[qb + d2], q1 = (float)qkv[qb + d2 + 64];
  float k0 = (float)qkv[qb + D_MODEL + d2], k1 = (float)qkv[qb + D_MODEL + d2 + 64];
  float v0 = (float)qkv[qb + 2 * D_MODEL + d2], v1 = (float)qkv[qb + 2 * D_MODEL + d2 + 64];

  float qr0 = q0 * c - q1 * s, qr1 = q1 * c + q0 * s;
  float kr0 = k0 * c - k1 * s, kr1 = k1 * c + k0 * s;

  size_t hb = ((size_t)(b * N_HEADS + h)) * SEQ_T;
  size_t qo = (hb + t) * HEAD_DIM;
  Q[qo + d2]       = (bf16)(qr0 * QK_SCALE);
  Q[qo + d2 + 64]  = (bf16)(qr1 * QK_SCALE);
  Kd[qo + d2]      = (bf16)kr0;
  Kd[qo + d2 + 64] = (bf16)kr1;
  size_t vb = ((size_t)(b * N_HEADS + h)) * HEAD_DIM;
  Vt[(vb + d2) * SEQ_T + t]      = (bf16)v0;
  Vt[(vb + d2 + 64) * SEQ_T + t] = (bf16)v1;
}

// ---------------------------------------------------------------------------
// 5) Flash attention (bidirectional). Block = 4 waves, wave = 16 query rows.
//    grid (T/64, B*H). Online softmax; P re-laid C->A layout via wave LDS.
// ---------------------------------------------------------------------------
__global__ void __launch_bounds__(128) attention_kernel(const bf16* __restrict__ Q,
                                                        const bf16* __restrict__ Kd,
                                                        const bf16* __restrict__ Vt,
                                                        bf16* __restrict__ AO) {
  __shared__ __align__(32) bf16 pls[4][16 * 32];
  const int tid  = threadIdx.x;
  const int wave = tid >> 5;
  const int lane = tid & 31;
  const int hi   = lane >> 4;
  const int lr   = lane & 15;

  const int bh = blockIdx.y;
  const int b  = bh >> 4;
  const int h  = bh & (N_HEADS - 1);
  const int q0 = blockIdx.x * 64 + wave * 16;

  const bf16* Qb = Q  + (size_t)bh * SEQ_T * HEAD_DIM;
  const bf16* Kb = Kd + (size_t)bh * SEQ_T * HEAD_DIM;
  const bf16* Vb = Vt + (size_t)bh * HEAD_DIM * SEQ_T;
  bf16* myp = &pls[wave][0];

  // Q in A-layout registers, 4 chunks of K-dim 32
  BV16 qa[4];
#pragma unroll
  for (int t = 0; t < 4; ++t) {
    const bf16* qp = Qb + (size_t)(q0 + lr) * HEAD_DIM + 32 * t + 16 * hi;
    qa[t].h[0] = *(const v8bf*)(qp);
    qa[t].h[1] = *(const v8bf*)(qp + 8);
  }

  v8f o[8];
#pragma unroll
  for (int c = 0; c < 8; ++c) o[c] = {};
  float m_i[8], l_i[8], alpha[8];
#pragma unroll
  for (int i = 0; i < 8; ++i) { m_i[i] = -1e30f; l_i[i] = 0.f; }

  for (int j = 0; j < SEQ_T / 32; ++j) {
    // ---- S = Q * K^T  (16 x 32 scores as two 16x16 C tiles) ----
    v8f s[2]; s[0] = {}; s[1] = {};
#pragma unroll
    for (int nt = 0; nt < 2; ++nt) {
      int krow = j * 32 + nt * 16 + lr;
#pragma unroll
      for (int t = 0; t < 4; ++t) {
        v16bf kb = *(const v16bf*)(Kb + (size_t)krow * HEAD_DIM + 32 * t + 16 * hi);
        s[nt] = wmma_bf16(qa[t].v, kb, s[nt]);
      }
    }
    // ---- online softmax (row groups: lanes 0-15 = rows 0-7, lanes 16-31 = rows 8-15) ----
#pragma unroll
    for (int i = 0; i < 8; ++i) {
      float s0 = s[0][i], s1 = s[1][i];
      float mx = fmaxf(s0, s1);
#pragma unroll
      for (int d = 8; d >= 1; d >>= 1) mx = fmaxf(mx, __shfl_xor(mx, d, 32));
      float mnew = fmaxf(m_i[i], mx);
      float al = __expf(m_i[i] - mnew);
      float p0 = __expf(s0 - mnew);
      float p1 = __expf(s1 - mnew);
      float rs = p0 + p1;
#pragma unroll
      for (int d = 8; d >= 1; d >>= 1) rs += __shfl_xor(rs, d, 32);
      l_i[i] = l_i[i] * al + rs;
      m_i[i] = mnew;
      alpha[i] = al;
      int r = i + 8 * hi;                  // C-layout row of this element
      myp[r * 32 + lr]      = (bf16)p0;
      myp[r * 32 + 16 + lr] = (bf16)p1;
    }
#pragma unroll
    for (int c = 0; c < 8; ++c)
#pragma unroll
      for (int i = 0; i < 8; ++i) o[c][i] *= alpha[i];

    asm volatile("s_wait_dscnt 0" ::: "memory");   // wave-private LDS bounce complete

    // P in A-layout (16x32)
    BV16 pa;
    pa.h[0] = *(const v8bf*)(myp + lr * 32 + 16 * hi);
    pa.h[1] = *(const v8bf*)(myp + lr * 32 + 16 * hi + 8);

    // ---- O += P * V  (V^T stored so B-operand loads are contiguous) ----
#pragma unroll
    for (int c = 0; c < 8; ++c) {
      v16bf vb = *(const v16bf*)(Vb + (size_t)(c * 16 + lr) * SEQ_T + j * 32 + 16 * hi);
      o[c] = wmma_bf16(pa.v, vb, o[c]);
    }
  }

  // ---- epilogue: normalize and scatter to [B,T,D] bf16 ----
#pragma unroll
  for (int i = 0; i < 8; ++i) {
    float rl = 1.f / l_i[i];
    int trow = q0 + i + 8 * hi;
    size_t base = ((size_t)(b * SEQ_T + trow)) * D_MODEL + (size_t)h * HEAD_DIM;
#pragma unroll
    for (int c = 0; c < 8; ++c)
      AO[base + c * 16 + lr] = (bf16)(o[c][i] * rl);
  }
}

// ---------------------------------------------------------------------------
// launcher
// ---------------------------------------------------------------------------
extern "C" void kernel_launch(void* const* d_in, const int* in_sizes, int n_in,
                              void* d_out, int out_size, void* d_ws, size_t ws_size,
                              hipStream_t stream) {
  const float* x      = (const float*)d_in[0];
  const float* norm_w = (const float*)d_in[1];
  const float* w_qkv  = (const float*)d_in[2];
  const float* w_out  = (const float*)d_in[3];
  float* out = (float*)d_out;

  // workspace carve-up (bf16 buffers), all offsets 256B-aligned
  char* w = (char*)d_ws;
  bf16* h_bf    = (bf16*)w; w += (size_t)NROWS * D_MODEL * 2;          // 16 MB
  bf16* wqkv_bf = (bf16*)w; w += (size_t)3 * D_MODEL * D_MODEL * 2;    // 24 MB
  bf16* wout_bf = (bf16*)w; w += (size_t)D_MODEL * D_MODEL * 2;        //  8 MB
  bf16* qkv_bf  = (bf16*)w; w += (size_t)NROWS * 3 * D_MODEL * 2;      // 48 MB
  bf16* q_bf    = (bf16*)w; w += (size_t)NROWS * D_MODEL * 2;          // 16 MB
  bf16* k_bf    = (bf16*)w; w += (size_t)NROWS * D_MODEL * 2;          // 16 MB
  bf16* vt_bf   = (bf16*)w; w += (size_t)NROWS * D_MODEL * 2;          // 16 MB
  bf16* ao_bf   = (bf16*)w;                                            // 16 MB

  // 1) weight casts
  cast_f32_bf16<<<2048, 256, 0, stream>>>(w_qkv, wqkv_bf, 3 * D_MODEL * D_MODEL);
  cast_f32_bf16<<<2048, 256, 0, stream>>>(w_out, wout_bf, D_MODEL * D_MODEL);

  // 2) RMSNorm
  rmsnorm_kernel<<<NROWS, 256, 0, stream>>>(x, norm_w, h_bf);

  // 3) QKV projection: [4096,2048] x [6144,2048]^T -> bf16 [4096,6144]
  gemm_bt_kernel<true><<<dim3(3 * D_MODEL / 128, NROWS / 128), 256, 0, stream>>>(
      h_bf, wqkv_bf, NROWS, 3 * D_MODEL, D_MODEL, qkv_bf, nullptr, nullptr);

  // 4) RoPE + split + V transpose
  rope_split_kernel<<<(BATCH * SEQ_T * N_HEADS * 64) / 256, 256, 0, stream>>>(
      qkv_bf, q_bf, k_bf, vt_bf);

  // 5) attention
  attention_kernel<<<dim3(SEQ_T / 64, BATCH * N_HEADS), 128, 0, stream>>>(
      q_bf, k_bf, vt_bf, ao_bf);

  // 6) output projection + residual -> f32 d_out
  gemm_bt_kernel<false><<<dim3(D_MODEL / 128, NROWS / 128), 256, 0, stream>>>(
      ao_bf, wout_bf, NROWS, D_MODEL, D_MODEL, nullptr, out, x);
}
